// TransformerBlock_11519102288410
// MI455X (gfx1250) — matrix operand order repaired and checked by reading, compiled, not run on hardware
//
#include <hip/hip_runtime.h>
#include <hip/hip_bf16.h>
#include <cstdint>
#include <cstddef>

// ---------------------------------------------------------------------------
// Types / helpers
// ---------------------------------------------------------------------------
typedef __bf16 bf16;
typedef bf16  v16bf __attribute__((ext_vector_type(16)));
typedef float v8f   __attribute__((ext_vector_type(8)));
typedef unsigned int u32x4 __attribute__((ext_vector_type(4)));
typedef int i32x4 __attribute__((ext_vector_type(4)));
typedef int i32x8 __attribute__((ext_vector_type(8)));

#define DEV static __device__ __forceinline__

DEV bf16 f2bf(float f) {
  unsigned u = __float_as_uint(f);
  unsigned r = (u + 0x7FFFu + ((u >> 16) & 1u)) >> 16;   // round-to-nearest-even
  unsigned short s = (unsigned short)r;
  return __builtin_bit_cast(bf16, s);
}
DEV v8f vzero8() { v8f z = {0.f,0.f,0.f,0.f,0.f,0.f,0.f,0.f}; return z; }

// DPP16 row rotate (within 16-lane rows) — pure VALU cross-lane, no LDS.
template <int N>  // rotate amount 1..15, ctrl = ROW_ROR(0x120 + N)
DEV float dpp_ror(float x) {
  int i = __builtin_bit_cast(int, x);
  int r = __builtin_amdgcn_update_dpp(i, i, 0x120 + N, 0xf, 0xf, false);
  return __builtin_bit_cast(float, r);
}
// all-reduce across a 16-lane row: every lane ends with the row result
DEV float row16_max(float x) {
  x = fmaxf(x, dpp_ror<8>(x));
  x = fmaxf(x, dpp_ror<4>(x));
  x = fmaxf(x, dpp_ror<2>(x));
  x = fmaxf(x, dpp_ror<1>(x));
  return x;
}
DEV float row16_sum(float x) {
  x += dpp_ror<8>(x);
  x += dpp_ror<4>(x);
  x += dpp_ror<2>(x);
  x += dpp_ror<1>(x);
  return x;
}

// Problem constants (match reference)
static constexpr int Bx = 4, Tt = 2048, Cc = 1024, Hh = 16, Dh = 64;
static constexpr int Mrows = Bx * Tt;          // 8192
static constexpr int HD = Hh * Dh;             // 1024

// ---------------------------------------------------------------------------
// TDM: 2-D tile load global -> LDS via Tensor Data Mover (D# per ISA 08_async)
//   tile_d0 elems (bf16) per row, tile_d1 rows, source row stride = stride_e.
//   LDS padding: 16 DWORDs data (+12 DWORDs pad) -> 112 B row pitch (LDT=56).
// ---------------------------------------------------------------------------
DEV void tdm_load_2d_bf16(unsigned lds_addr, const bf16* gsrc,
                          unsigned tile_d0, unsigned tile_d1,
                          unsigned long long stride_e) {
  unsigned long long ga = (unsigned long long)(uintptr_t)gsrc;
  u32x4 g0;
  g0[0] = 1u;                                           // count=1, user mode
  g0[1] = lds_addr;                                     // LDS byte address
  g0[2] = (unsigned)(ga & 0xFFFFFFFFull);               // global addr lo
  g0[3] = (unsigned)((ga >> 32) & 0x1FFFFFFull) | (2u << 30);  // addr hi | type=2
  i32x8 g1;
  g1[0] = (int)((1u << 16)      // data_size = 2 bytes
              | (1u << 20)      // pad_enable
              | (3u << 22)      // pad_interval: 16 DWORDs between pads
              | (11u << 25));   // pad_amount: 12 DWORDs
  g1[1] = (int)((tile_d0 & 0xFFFFu) << 16);             // tensor_dim0[15:0]
  g1[2] = (int)(((tile_d0 >> 16) & 0xFFFFu) | ((tile_d1 & 0xFFFFu) << 16));
  g1[3] = (int)(((tile_d1 >> 16) & 0xFFFFu) | ((tile_d0 & 0xFFFFu) << 16)); // | tile_dim0
  g1[4] = (int)(tile_d1 & 0xFFFFu);                     // tile_dim1 (tile_dim2=0)
  g1[5] = (int)(stride_e & 0xFFFFFFFFull);              // tensor_dim0_stride lo
  g1[6] = (int)((stride_e >> 32) & 0xFFFFull);          // stride hi (dim1_stride=0)
  g1[7] = 0;
  i32x4 gz = {0, 0, 0, 0};                              // groups 2/3 unused (2-D)
#if defined(__clang_major__) && (__clang_major__ >= 23)
  i32x8 gz8 = {0, 0, 0, 0, 0, 0, 0, 0};
  __builtin_amdgcn_tensor_load_to_lds(g0, g1, gz, gz, gz8, 0);
#else
  __builtin_amdgcn_tensor_load_to_lds(g0, g1, gz, gz, 0);
#endif
}

// ---------------------------------------------------------------------------
// fp32 -> bf16 with transpose: in [R][Cn] f32 -> out [Cn][R] bf16
// 2-D grid: x covers columns (coalesced read), y covers rows.
// ---------------------------------------------------------------------------
__global__ void cvtT_kernel(const float* __restrict__ in, bf16* __restrict__ out,
                            int R, int Cn) {
  const int c = blockIdx.x * blockDim.x + threadIdx.x;
  const int r = blockIdx.y;
  if (c < Cn) out[(size_t)c * R + r] = f2bf(in[(size_t)r * Cn + c]);
}

// ---------------------------------------------------------------------------
// LayerNorm over C=1024, one row per 256-thread block, bf16 output
// ---------------------------------------------------------------------------
__global__ __launch_bounds__(256)
void ln_bf16_kernel(const float* __restrict__ x, const float* __restrict__ g,
                    const float* __restrict__ be, bf16* __restrict__ out) {
  const int C = Cc;
  const int row = blockIdx.x;
  const int tid = threadIdx.x;
  const float* xr = x + (size_t)row * C;
  __shared__ float red[8];

  float v[4]; float s = 0.f;
#pragma unroll
  for (int j = 0; j < 4; ++j) { v[j] = xr[tid + j * 256]; s += v[j]; }
  s = row16_sum(s);
  s += __shfl_xor(s, 16, 32);          // combine the two 16-lane halves
  if ((tid & 31) == 0) red[tid >> 5] = s;
  __syncthreads();
  float tot = 0.f;
#pragma unroll
  for (int w = 0; w < 8; ++w) tot += red[w];
  const float mean = tot * (1.0f / C);

  float s2 = 0.f;
#pragma unroll
  for (int j = 0; j < 4; ++j) { float d = v[j] - mean; s2 += d * d; }
  s2 = row16_sum(s2);
  s2 += __shfl_xor(s2, 16, 32);
  __syncthreads();
  if ((tid & 31) == 0) red[tid >> 5] = s2;
  __syncthreads();
  float tv = 0.f;
#pragma unroll
  for (int w = 0; w < 8; ++w) tv += red[w];
  const float rstd = rsqrtf(tv * (1.0f / C) + 1e-5f);

  bf16* orow = out + (size_t)row * C;
#pragma unroll
  for (int j = 0; j < 4; ++j) {
    int c = tid + j * 256;
    orow[c] = f2bf((v[j] - mean) * rstd * g[c] + be[c]);
  }
}

// ---------------------------------------------------------------------------
// Tiled WMMA GEMM: C[M,N](f32 acc) = A[M,K]_bf16 * Bt[N,K]_bf16 (B pre-transposed)
// block tile 64x128, K-step 32, 8 waves (2x4), each wave 32x32 (4 WMMA tiles).
// Tiles streamed global->LDS by the Tensor Data Mover, double-buffered so the
// TDM copies tile i+1 while the waves compute on tile i.
// ---------------------------------------------------------------------------
static constexpr int BM = 64, BN = 128, BK = 32;
static constexpr int LDT = 56;   // LDS row pitch in halves: 112 B (16 DW + 12 DW TDM pad)

template <bool HAS_BIAS, bool RELU, bool HAS_RES, bool OUT_F32, bool OUT_BF16, bool OUT_T>
__global__ __launch_bounds__(256)
void gemm_bf16_kernel(const bf16* __restrict__ A, const bf16* __restrict__ Bt,
                      const float* __restrict__ bias, const float* __restrict__ res,
                      float* __restrict__ outF, bf16* __restrict__ outB,
                      int M, int N, int K) {
  __shared__ bf16 As[2][BM * LDT];
  __shared__ bf16 Bs[2][BN * LDT];   // Bs[n][k]

  const int tid  = threadIdx.x;
  const int lane = tid & 31;
  const int wave = tid >> 5;
  const int wm = wave >> 2;       // 0..1 (M direction)
  const int wn = wave & 3;        // 0..3 (N direction)
  const int n16 = lane & 15, hi = lane >> 4;
  const int m0 = blockIdx.y * BM;
  const int n0 = blockIdx.x * BN;

  v8f acc[2][2];
#pragma unroll
  for (int i = 0; i < 2; ++i)
#pragma unroll
    for (int j = 0; j < 2; ++j) acc[i][j] = vzero8();

  const bf16* Arow = A  + (size_t)m0 * K;
  const bf16* Brow = Bt + (size_t)n0 * K;
  const int nt = K / BK;

  if (wave == 0) {   // prime the pipeline: tile 0 into buffer 0
    tdm_load_2d_bf16((unsigned)(uintptr_t)&As[0][0], Arow, BK, BM, (unsigned long long)K);
    tdm_load_2d_bf16((unsigned)(uintptr_t)&Bs[0][0], Brow, BK, BN, (unsigned long long)K);
  }

  for (int it = 0; it < nt; ++it) {
    const int cur = it & 1;
    if (wave == 0) {
      if (it + 1 < nt) {   // stream next tile into the other buffer, then wait for current
        const int kn = (it + 1) * BK;
        tdm_load_2d_bf16((unsigned)(uintptr_t)&As[cur ^ 1][0], Arow + kn, BK, BM, (unsigned long long)K);
        tdm_load_2d_bf16((unsigned)(uintptr_t)&Bs[cur ^ 1][0], Brow + kn, BK, BN, (unsigned long long)K);
        __builtin_amdgcn_s_wait_tensorcnt(2);   // in-order TDM: <=2 left => tile `it` landed
      } else {
        __builtin_amdgcn_s_wait_tensorcnt(0);
      }
    }
    __syncthreads();

    // A fragments (16x32 bf16; K contiguous per lane half, k = 16*hi + s, matching B)
    v16bf a[2], bq[2];
#pragma unroll
    for (int i = 0; i < 2; ++i) {
      const bf16* p = &As[cur][(wm * 32 + i * 16 + n16) * LDT];
      const int b1 = hi * 16;
#pragma unroll
      for (int j = 0; j < 8; ++j) { a[i][j] = p[b1 + j]; a[i][8 + j] = p[b1 + 8 + j]; }
    }
    // B fragments (32x16: lane half selects K 0-15 / 16-31, contiguous in Bs)
#pragma unroll
    for (int i = 0; i < 2; ++i) {
      const bf16* p = &Bs[cur][(wn * 32 + i * 16 + n16) * LDT + hi * 16];
#pragma unroll
      for (int j = 0; j < 16; ++j) bq[i][j] = p[j];
    }
#pragma unroll
    for (int i = 0; i < 2; ++i)
#pragma unroll
      for (int j = 0; j < 2; ++j)
        acc[i][j] = __builtin_amdgcn_wmma_f32_16x16x32_bf16(
            false, a[i], false, bq[j], (short)0, acc[i][j], false, false);
    __syncthreads();   // all waves done with buf `cur` before TDM reuses it
  }

  // epilogue (C/D layout: row = v + 8*hi, col = lane&15)
#pragma unroll
  for (int i = 0; i < 2; ++i)
#pragma unroll
    for (int j = 0; j < 2; ++j) {
      const int col = n0 + wn * 32 + j * 16 + n16;
      const float bval = HAS_BIAS ? bias[col] : 0.0f;
#pragma unroll
      for (int v = 0; v < 8; ++v) {
        const int row = m0 + wm * 32 + i * 16 + v + 8 * hi;
        float val = acc[i][j][v];
        if (HAS_BIAS) val += bval;
        if (RELU)     val = fmaxf(val, 0.0f);
        if (HAS_RES)  val += res[(size_t)row * N + col];
        if (OUT_F32)  outF[(size_t)row * N + col] = val;
        if (OUT_BF16) {
          if (OUT_T) {
            const int bb = row / Tt, t = row - bb * Tt;   // transposed per batch
            outB[((size_t)bb * N + col) * Tt + t] = f2bf(val);
          } else {
            outB[(size_t)row * N + col] = f2bf(val);
          }
        }
      }
    }
}

// ---------------------------------------------------------------------------
// Flash attention (causal). One wave owns a 16-query tile of one (b,h).
// Q in registers as A-frags; K B-frags and V^T B-frags straight from global
// (both K-contiguous per lane); P re-laid-out D->A via a small LDS bounce.
// Row-wise softmax reductions use DPP16 rotations (no LDS round-trips).
// ---------------------------------------------------------------------------
static constexpr int ALD = 36;  // LDS stride (halves)

__global__ __launch_bounds__(256)
void attn_kernel(const bf16* __restrict__ q, const bf16* __restrict__ k,
                 const bf16* __restrict__ vT,   // [B][H][64][T]
                 bf16* __restrict__ o) {
  __shared__ bf16 Ps[8][16][ALD];   // [wave][qrow][key]

  const int tid = threadIdx.x, lane = tid & 31, wv = tid >> 5;
  const int wid = blockIdx.x * 8 + wv;
  const int qtiles = Tt / 16;
  const int qt = wid % qtiles;
  const int h  = (wid / qtiles) % Hh;
  const int b  = wid / (qtiles * Hh);
  const int q0 = qt * 16;
  const int n16 = lane & 15, hi = lane >> 4;
  const size_t baseBT = (size_t)b * Tt;
  const int C = HD;

  // Q A-fragments: frag f covers dims [f*32, f*32+32)
  v16bf aq[2];
  {
    const bf16* qr = q + (baseBT + q0 + n16) * C + h * 64;
    const int b1 = hi * 16;
#pragma unroll
    for (int f = 0; f < 2; ++f)
#pragma unroll
      for (int j = 0; j < 8; ++j) {
        aq[f][j]     = qr[f * 32 + b1 + j];
        aq[f][8 + j] = qr[f * 32 + b1 + 8 + j];
      }
  }

  v8f oacc[4];
#pragma unroll
  for (int g = 0; g < 4; ++g) oacc[g] = vzero8();
  float mrow[8], lrow[8];
#pragma unroll
  for (int i = 0; i < 8; ++i) { mrow[i] = -__builtin_inff(); lrow[i] = 0.f; }

  const int ktiles = (q0 + 16 + 31) / 32;
  const int kb = hi * 16;
  const bf16* vTh = vT + ((size_t)b * Hh + h) * 64 * Tt;

  for (int t = 0; t < ktiles; ++t) {
    const int k0 = t * 32;

    // prefetch next key tile (K rows + V^T rows) one iteration ahead
    if (t + 1 < ktiles) {
      __builtin_prefetch(k + (baseBT + k0 + 32 + lane) * C + h * 64, 0, 1);
      __builtin_prefetch(vTh + (size_t)(lane << 1) * Tt + k0 + 32, 0, 1);
    }

    // S = Q K^T for two 16-key groups (B frags straight from global K rows)
    v8f s0 = vzero8(), s1 = vzero8();
#pragma unroll
    for (int f = 0; f < 2; ++f) {
      v16bf bk0, bk1;
      const bf16* kr0 = k + (baseBT + k0 + n16) * C + h * 64 + f * 32 + kb;
      const bf16* kr1 = k + (baseBT + k0 + 16 + n16) * C + h * 64 + f * 32 + kb;
#pragma unroll
      for (int j = 0; j < 16; ++j) { bk0[j] = kr0[j]; bk1[j] = kr1[j]; }
      s0 = __builtin_amdgcn_wmma_f32_16x16x32_bf16(false, aq[f], false, bk0, (short)0, s0, false, false);
      s1 = __builtin_amdgcn_wmma_f32_16x16x32_bf16(false, aq[f], false, bk1, (short)0, s1, false, false);
    }

    // online softmax (rows live as v + 8*hi across 16-lane rows)
    const float sc = 0.125f;  // 1/sqrt(64)
    const int c0 = k0 + n16, c1 = c0 + 16;
#pragma unroll
    for (int vv = 0; vv < 8; ++vv) {
      const int rowg = q0 + vv + 8 * hi;
      float x0 = (c0 <= rowg) ? s0[vv] * sc : -__builtin_inff();
      float x1 = (c1 <= rowg) ? s1[vv] * sc : -__builtin_inff();
      const float mx = row16_max(fmaxf(x0, x1));
      const float mn = fmaxf(mrow[vv], mx);
      const float p0 = __expf(x0 - mn);
      const float p1 = __expf(x1 - mn);
      const float rs = row16_sum(p0 + p1);
      const float al = __expf(mrow[vv] - mn);
      lrow[vv] = lrow[vv] * al + rs;
      mrow[vv] = mn;
#pragma unroll
      for (int g = 0; g < 4; ++g) oacc[g][vv] *= al;
      Ps[wv][vv + 8 * hi][n16]      = f2bf(p0);
      Ps[wv][vv + 8 * hi][16 + n16] = f2bf(p1);
    }

    // P (16x32) as A-frag from LDS; V^T B-frags from global; O += P*V
    v16bf pa;
    {
      const bf16* pr = &Ps[wv][n16][0];
      const int b1 = hi * 16;
#pragma unroll
      for (int j = 0; j < 8; ++j) { pa[j] = pr[b1 + j]; pa[8 + j] = pr[b1 + 8 + j]; }
    }
#pragma unroll
    for (int g = 0; g < 4; ++g) {
      v16bf vb;
      const bf16* vp = vTh + (size_t)(g * 16 + n16) * Tt + k0 + kb;
#pragma unroll
      for (int j = 0; j < 16; ++j) vb[j] = vp[j];
      oacc[g] = __builtin_amdgcn_wmma_f32_16x16x32_bf16(false, pa, false, vb, (short)0, oacc[g], false, false);
    }
  }

  // normalize and store bf16 output [B*T, H*Dh]
#pragma unroll
  for (int vv = 0; vv < 8; ++vv) {
    const float inv = 1.0f / lrow[vv];
    const int rowg = q0 + vv + 8 * hi;
    bf16* orow = o + (baseBT + rowg) * C + h * 64;
#pragma unroll
    for (int g = 0; g < 4; ++g) orow[g * 16 + n16] = f2bf(oacc[g][vv] * inv);
  }
}

// ---------------------------------------------------------------------------
// Host-side launch
// ---------------------------------------------------------------------------
extern "C" void kernel_launch(void* const* d_in, const int* in_sizes, int n_in,
                              void* d_out, int out_size, void* d_ws, size_t ws_size,
                              hipStream_t stream) {
  const float* x  = (const float*)d_in[0];
  const float* Wq = (const float*)d_in[1];
  const float* Wk = (const float*)d_in[2];
  const float* Wv = (const float*)d_in[3];
  const float* Wo = (const float*)d_in[4];
  const float* bo = (const float*)d_in[5];
  const float* W1 = (const float*)d_in[6];
  const float* b1 = (const float*)d_in[7];
  const float* W2 = (const float*)d_in[8];
  const float* b2 = (const float*)d_in[9];
  const float* g1 = (const float*)d_in[10];
  const float* be1 = (const float*)d_in[11];
  const float* g2 = (const float*)d_in[12];
  const float* be2 = (const float*)d_in[13];
  float* out = (float*)d_out;

  char* ws = (char*)d_ws;
  size_t off = 0;
  auto alloc = [&](size_t bytes) -> char* {
    char* p = ws + off;
    off += (bytes + 255) & ~(size_t)255;
    return p;
  };
  const size_t nW   = (size_t)Cc * HD;         // 1M
  const size_t nW1  = (size_t)Cc * 4 * Cc;     // 4M
  const size_t nAct = (size_t)Mrows * Cc;      // 8M
  const size_t nMid = (size_t)Mrows * 4 * Cc;  // 32M

  bf16* WqT = (bf16*)alloc(nW * 2);    // [HD][C]
  bf16* WkT = (bf16*)alloc(nW * 2);
  bf16* WvT = (bf16*)alloc(nW * 2);
  bf16* WoT = (bf16*)alloc(nW * 2);    // [C][HD]
  bf16* W1T = (bf16*)alloc(nW1 * 2);   // [4C][C]
  bf16* W2T = (bf16*)alloc(nW1 * 2);   // [C][4C]
  bf16* hB  = (bf16*)alloc(nAct * 2);  // LN output (reused for LN2)
  bf16* qB  = (bf16*)alloc(nAct * 2);
  bf16* kB  = (bf16*)alloc(nAct * 2);
  bf16* vTB = (bf16*)alloc(nAct * 2);  // V transposed: [B][H][64][T]
  bf16* aB  = (bf16*)alloc(nAct * 2);  // attention output
  bf16* midB = (bf16*)alloc(nMid * 2); // MLP hidden
  float* x1 = (float*)alloc(nAct * 4); // post-attention residual stream
  (void)ws_size; (void)n_in; (void)in_sizes; (void)out_size;

  // weight convert + transpose (once per launch)
  cvtT_kernel<<<dim3(HD / 256, Cc), 256, 0, stream>>>(Wq, WqT, Cc, HD);
  cvtT_kernel<<<dim3(HD / 256, Cc), 256, 0, stream>>>(Wk, WkT, Cc, HD);
  cvtT_kernel<<<dim3(HD / 256, Cc), 256, 0, stream>>>(Wv, WvT, Cc, HD);
  cvtT_kernel<<<dim3(Cc / 256, HD), 256, 0, stream>>>(Wo, WoT, HD, Cc);
  cvtT_kernel<<<dim3(4 * Cc / 256, Cc), 256, 0, stream>>>(W1, W1T, Cc, 4 * Cc);
  cvtT_kernel<<<dim3(Cc / 256, 4 * Cc), 256, 0, stream>>>(W2, W2T, 4 * Cc, Cc);

  // LN1
  ln_bf16_kernel<<<Mrows, 256, 0, stream>>>(x, g1, be1, hB);

  // QKV projections (M=8192, N=1024, K=1024); V written transposed per (b,h)
  dim3 gQKV(HD / BN, Mrows / BM);
  gemm_bf16_kernel<false,false,false,false,true,false><<<gQKV, 256, 0, stream>>>(
      hB, WqT, nullptr, nullptr, nullptr, qB, Mrows, HD, Cc);
  gemm_bf16_kernel<false,false,false,false,true,false><<<gQKV, 256, 0, stream>>>(
      hB, WkT, nullptr, nullptr, nullptr, kB, Mrows, HD, Cc);
  gemm_bf16_kernel<false,false,false,false,true,true><<<gQKV, 256, 0, stream>>>(
      hB, WvT, nullptr, nullptr, nullptr, vTB, Mrows, HD, Cc);

  // attention: B*H*(T/16) waves / 8 waves per block
  attn_kernel<<<(Bx * Hh * (Tt / 16)) / 8, 256, 0, stream>>>(qB, kB, vTB, aB);

  // output projection + bias + residual -> x1 (f32)
  gemm_bf16_kernel<true,false,true,true,false,false><<<dim3(Cc / BN, Mrows / BM), 256, 0, stream>>>(
      aB, WoT, bo, x, x1, nullptr, Mrows, Cc, HD);

  // LN2
  ln_bf16_kernel<<<Mrows, 256, 0, stream>>>(x1, g2, be2, hB);

  // MLP up + bias + ReLU (M=8192, N=4096, K=1024) -> bf16
  gemm_bf16_kernel<true,true,false,false,true,false><<<dim3(4 * Cc / BN, Mrows / BM), 256, 0, stream>>>(
      hB, W1T, b1, nullptr, nullptr, midB, Mrows, 4 * Cc, Cc);

  // MLP down + bias + residual (M=8192, N=1024, K=4096) -> d_out (f32)
  gemm_bf16_kernel<true,false,true,true,false,false><<<dim3(Cc / BN, Mrows / BM), 256, 0, stream>>>(
      midB, W2T, b2, x1, out, nullptr, Mrows, Cc, 4 * Cc);
}
